// GaugeDynamicsEager_19464791786181
// MI455X (gfx1250) — matrix operand order repaired and checked by reading, not compiled, not run on hardware
//
#include <hip/hip_runtime.h>
#include <hip/hip_bf16.h>
#include <math.h>

// ===========================================================================
// GaugeDynamics (L2HMC) on MI455X / gfx1250.
//
// Roofline: 200 sequential net calls; each streams ~112MB of bf16-packed
// weights. The per-direction weight set (112MB) fits in the 192MB L2, so the
// chain is L2-bandwidth bound after first touch -> maximize wave parallelism
// and memory-level parallelism:
//   * weights pre-swizzled once into WMMA B-fragment layout (coalesced b128)
//   * K-split=8 partial-sum GEMM (grid N/128 x 8) + tiny reduce/act epilogue
//   * A-fragments staged in LDS per block (kills 8x redundant global reads)
//   * B-fragment register double-buffering (peeled tail -> no zero-fill,
//     no in-loop branch) + global_prefetch_b8 lookahead
// Workspace requirement: ~250 MB.
// ===========================================================================

typedef __bf16 v16bf __attribute__((ext_vector_type(16)));
typedef float  v8f   __attribute__((ext_vector_type(8)));

static __device__ inline unsigned short f2bf(float f) {
  union { __bf16 b; unsigned short u; } c;
  c.b = (__bf16)f;   // RNE conversion
  return c.u;
}

// ---------------------------------------------------------------------------
// Pack f32 activation matrix (M=64 x K, row-major) into WMMA A-fragments.
// Layout: frag[(mt*KT + kt)*32 + lane][16 bf16].
// A 16x32 bf16 layout: lanes 0-15 = rows, elems 0..7 -> K+0..7,
// elems 8..15 -> K+16..23; lanes 16-31 same rows, K offset +8.
// ---------------------------------------------------------------------------
__global__ __launch_bounds__(256)
void pack_a_kernel(const float* __restrict__ src, unsigned short* __restrict__ dst, int K)
{
  int tid = blockIdx.x * 256 + threadIdx.x;
  int KT  = K >> 5;
  if (tid >= 4 * KT * 32) return;
  int lane = tid & 31;
  int frag = tid >> 5;
  int mt = frag / KT;
  int kt = frag - mt * KT;
  int row = (mt << 4) + (lane & 15);
  int kb  = (kt << 5) + ((lane >> 4) << 3);          // +8 for upper half-wave
  const float* s = src + (size_t)row * K + kb;
  unsigned short* d = dst + (size_t)tid * 16;
#pragma unroll
  for (int e = 0; e < 8; ++e) d[e]     = f2bf(s[e]);
#pragma unroll
  for (int e = 0; e < 8; ++e) d[8 + e] = f2bf(s[16 + e]);
}

// ---------------------------------------------------------------------------
// Pack f32 weight matrix (K x N, row-major) into WMMA B-fragments.
// Layout: frag[(nt*KT + kt)*32 + lane][16 bf16].
// B 32x16 bf16 layout: col = lane%16; lanes 0-15: elems 0..7 -> K+0..7,
// elems 8..15 -> K+16..23; lanes 16-31 same, K offset +8.
// ---------------------------------------------------------------------------
__global__ __launch_bounds__(256)
void pack_w_kernel(const float* __restrict__ src, unsigned short* __restrict__ dst,
                   int K, int N)
{
  size_t tid = (size_t)blockIdx.x * 256 + threadIdx.x;
  int KT = K >> 5, NT = N >> 4;
  if (tid >= (size_t)NT * KT * 32) return;
  int lane = (int)(tid & 31);
  size_t frag = tid >> 5;
  int nt = (int)(frag / KT);
  int kt = (int)(frag - (size_t)nt * KT);
  int n  = (nt << 4) + (lane & 15);
  int kb = (kt << 5) + ((lane >> 4) << 3);           // +8 for upper half-wave
  const float* s = src + (size_t)kb * N + n;
  unsigned short* d = dst + tid * 16;
#pragma unroll
  for (int e = 0; e < 16; ++e) d[e] = f2bf(s[(size_t)(e < 8 ? e : e + 8) * N]);
}

// ---------------------------------------------------------------------------
// Partial GEMM: part[slot](64 x N) = A(64 x Kchunk) @ W(Kchunk x N).
// grid = (N/128, KSPLIT); block = 8 waves; wave owns one 16-col strip x
// four 16-row tiles. A chunk staged in LDS (4*KTB KB); weight stream is
// double-buffered in registers (tail-peeled) + prefetched.
// ---------------------------------------------------------------------------
__global__ __launch_bounds__(256)
void gemm_bf16_kernel(const unsigned short* __restrict__ Apk,
                      const unsigned short* __restrict__ Wpk,
                      float* __restrict__ part, int slotOff,
                      int N, int KT, int KTB)
{
  extern __shared__ char smem_raw[];
  const int kt0 = blockIdx.y * KTB;

  // ---- cooperative stage of A chunk into LDS (same frag-major layout) ----
  {
    uint4*       al = reinterpret_cast<uint4*>(smem_raw);
    const uint4* ag = reinterpret_cast<const uint4*>(Apk);
    const int units = 4 * KTB * 64;                  // 16B units (64 per 1KB frag)
    for (int u = threadIdx.x; u < units; u += 256) {
      int frag = u >> 6;
      int unit = u & 63;
      int mt   = frag / KTB;
      int ktl  = frag - mt * KTB;
      al[u] = ag[((size_t)(mt * KT + kt0 + ktl) << 6) + unit];
    }
  }
  __syncthreads();

  const int lane = threadIdx.x & 31;
  const int wave = threadIdx.x >> 5;
  const int nt   = blockIdx.x * 8 + wave;

  const v16bf* wf = reinterpret_cast<const v16bf*>(Wpk) +
                    ((size_t)nt * KT + kt0) * 32 + lane;
  const v16bf* al = reinterpret_cast<const v16bf*>(smem_raw) + lane;

  v8f acc0 = {}, acc1 = {}, acc2 = {}, acc3 = {};

  // steady state: unconditional next-fragment load (tail peeled below)
  v16bf b = wf[0];
  int ktl = 0;
  for (; ktl < KTB - 1; ++ktl) {
    __builtin_prefetch((const void*)(wf + (size_t)(ktl + 8) * 32), 0, 3);
    v16bf bn = wf[(size_t)(ktl + 1) * 32];                // next weight frag
    v16bf a0 = al[(size_t)(0 * KTB + ktl) * 32];          // A from LDS
    v16bf a1 = al[(size_t)(1 * KTB + ktl) * 32];
    v16bf a2 = al[(size_t)(2 * KTB + ktl) * 32];
    v16bf a3 = al[(size_t)(3 * KTB + ktl) * 32];
    acc0 = __builtin_amdgcn_wmma_f32_16x16x32_bf16(false, a0, false, b, (short)0, acc0, false, false);
    acc1 = __builtin_amdgcn_wmma_f32_16x16x32_bf16(false, a1, false, b, (short)0, acc1, false, false);
    acc2 = __builtin_amdgcn_wmma_f32_16x16x32_bf16(false, a2, false, b, (short)0, acc2, false, false);
    acc3 = __builtin_amdgcn_wmma_f32_16x16x32_bf16(false, a3, false, b, (short)0, acc3, false, false);
    b = bn;
  }
  {
    v16bf a0 = al[(size_t)(0 * KTB + ktl) * 32];
    v16bf a1 = al[(size_t)(1 * KTB + ktl) * 32];
    v16bf a2 = al[(size_t)(2 * KTB + ktl) * 32];
    v16bf a3 = al[(size_t)(3 * KTB + ktl) * 32];
    acc0 = __builtin_amdgcn_wmma_f32_16x16x32_bf16(false, a0, false, b, (short)0, acc0, false, false);
    acc1 = __builtin_amdgcn_wmma_f32_16x16x32_bf16(false, a1, false, b, (short)0, acc1, false, false);
    acc2 = __builtin_amdgcn_wmma_f32_16x16x32_bf16(false, a2, false, b, (short)0, acc2, false, false);
    acc3 = __builtin_amdgcn_wmma_f32_16x16x32_bf16(false, a3, false, b, (short)0, acc3, false, false);
  }

  // D layout: lane l -> col = l%16 ; VGPR r -> row r (lanes<16) / r+8 (>=16)
  float* o = part + (size_t)(slotOff + blockIdx.y) * 64 * N;
  const int n  = (nt << 4) + (lane & 15);
  const int rb = (lane < 16) ? 0 : 8;
  auto store_tile = [&](const v8f& a, int mt) {
#pragma unroll
    for (int r = 0; r < 8; ++r)
      o[(size_t)(mt * 16 + rb + r) * N + n] = a[r];
  };
  store_tile(acc0, 0); store_tile(acc1, 1); store_tile(acc2, 2); store_tile(acc3, 3);
}

// ---------------------------------------------------------------------------
// Epilogue: out = act( sum_p part[p] + bias ).  total = 64*N elements.
// ---------------------------------------------------------------------------
template<bool BIAS, bool RELU>
__global__ __launch_bounds__(256)
void epilogue_kernel(const float* __restrict__ part, int nparts, int N,
                     const float* __restrict__ bias, float* __restrict__ out,
                     int total)
{
  int i = blockIdx.x * 256 + threadIdx.x;
  if (i >= total) return;
  float s = 0.0f;
  for (int p = 0; p < nparts; ++p) s += part[(size_t)p * total + i];
  if (BIAS) s += bias[i % N];
  if (RELU) s = fmaxf(s, 0.0f);
  out[i] = s;
}

// ---------------------------------------------------------------------------
// Small elementwise kernels
// ---------------------------------------------------------------------------
__global__ __launch_bounds__(256)
void bias1_kernel(const float* __restrict__ b1n, const float* __restrict__ Wtn,
                  const float* __restrict__ ts, int step, float* __restrict__ bias, int H)
{
  int j = blockIdx.x * 256 + threadIdx.x;
  if (j < H) {
    float t0 = ts[2 * step], t1 = ts[2 * step + 1];
    bias[j] = b1n[j] + t0 * Wtn[j] + t1 * Wtn[H + j];
  }
}

__global__ __launch_bounds__(256)
void grad_kernel(const float* __restrict__ x, float* __restrict__ g, int n)
{
  int i = blockIdx.x * 256 + threadIdx.x;
  if (i < n) g[i] = sinf(x[i]);
}

__global__ __launch_bounds__(256)
void maskmul_kernel(const float* __restrict__ pos, const float* __restrict__ mask,
                    float* __restrict__ outb, int D, int n, int swap)
{
  int i = blockIdx.x * 256 + threadIdx.x;
  if (i < n) {
    float m = mask[i % D];
    if (swap) m = 1.0f - m;
    outb[i] = m * pos[i];
  }
}

// ---------------------------------------------------------------------------
// Leapfrog update kernels: one block per batch row; LDS reduce for sumlogdet.
// ---------------------------------------------------------------------------
__global__ __launch_bounds__(256)
void mom_update_kernel(float* __restrict__ mom, const float* __restrict__ g,
                       const float* __restrict__ yS, const float* __restrict__ yT,
                       const float* __restrict__ yQ,
                       const float* __restrict__ bsv, const float* __restrict__ csv,
                       const float* __restrict__ btrv,
                       const float* __restrict__ bqv, const float* __restrict__ cqv,
                       const float* __restrict__ epsp, float* __restrict__ sld,
                       int D, int backward)
{
  __shared__ float red[256];
  const int b = blockIdx.x;
  const float e = epsp[0];
  float lsum = 0.0f;
  for (int d = threadIdx.x; d < D; d += 256) {
    size_t i = (size_t)b * D + d;
    float S  = tanhf(yS[i] + bsv[d]) * expf(csv[d]);
    float T  = yT[i] + btrv[d];
    float Q  = tanhf(yQ[i] + bqv[d]) * expf(cqv[d]);
    float sc = (backward ? -0.5f : 0.5f) * e * S;
    float tr = e * Q;
    float gv = g[i], mv = mom[i];
    float nm = backward ? expf(sc) * (mv + 0.5f * e * (expf(tr) * gv - T))
                        : mv * expf(sc) - 0.5f * e * (expf(tr) * gv - T);
    mom[i] = nm;
    lsum += sc;
  }
  red[threadIdx.x] = lsum; __syncthreads();
  for (int s = 128; s; s >>= 1) {
    if (threadIdx.x < s) red[threadIdx.x] += red[threadIdx.x + s];
    __syncthreads();
  }
  if (threadIdx.x == 0) sld[b] += red[0];
}

__global__ __launch_bounds__(256)
void pos_update_kernel(float* __restrict__ pos, const float* __restrict__ mom,
                       const float* __restrict__ mask,
                       const float* __restrict__ yS, const float* __restrict__ yT,
                       const float* __restrict__ yQ,
                       const float* __restrict__ bsv, const float* __restrict__ csv,
                       const float* __restrict__ btrv,
                       const float* __restrict__ bqv, const float* __restrict__ cqv,
                       const float* __restrict__ epsp, float* __restrict__ sld,
                       int D, int swap, int backward)
{
  __shared__ float red[256];
  const int b = blockIdx.x;
  const float e = epsp[0];
  float lsum = 0.0f;
  for (int d = threadIdx.x; d < D; d += 256) {
    size_t i = (size_t)b * D + d;
    float m = mask[d]; if (swap) m = 1.0f - m;
    float minv = 1.0f - m;
    float S  = tanhf(yS[i] + bsv[d]) * expf(csv[d]);
    float T  = yT[i] + btrv[d];
    float Q  = tanhf(yQ[i] + bqv[d]) * expf(cqv[d]);
    float sc = e * S, tr = e * Q;
    float p = pos[i], v = mom[i];
    float np = backward ? m * p + minv * expf(sc) * (p - e * (expf(tr) * v + T))
                        : m * p + minv * (p * expf(sc) + e * (expf(tr) * v + T));
    pos[i] = np;
    lsum += minv * sc;
  }
  red[threadIdx.x] = lsum; __syncthreads();
  for (int s = 128; s; s >>= 1) {
    if (threadIdx.x < s) red[threadIdx.x] += red[threadIdx.x + s];
    __syncthreads();
  }
  if (threadIdx.x == 0) sld[b] += red[0];
}

// ---------------------------------------------------------------------------
// Final accept/combine kernel: one block per batch row.
// out = [position_post (B*D) | momentum_post (B*D) | ap (B) | position_out (B*D)]
// ---------------------------------------------------------------------------
__global__ __launch_bounds__(256)
void finalize_kernel(const float* __restrict__ pos0,
                     const float* __restrict__ m0f, const float* __restrict__ m0b,
                     const float* __restrict__ pf,  const float* __restrict__ mf,
                     const float* __restrict__ pb,  const float* __restrict__ mb,
                     const float* __restrict__ sldf, const float* __restrict__ sldb,
                     const float* __restrict__ u_dir, const float* __restrict__ u_accept,
                     float* __restrict__ out, int Bn, int D)
{
  __shared__ float red[256];
  __shared__ float sh[2];
  const int b = blockIdx.x;
  float p0 = 0, kf0 = 0, kb0 = 0, ppf = 0, kmf = 0, ppb = 0, kmb = 0;
  for (int d = threadIdx.x; d < D; d += 256) {
    size_t i = (size_t)b * D + d;
    p0  += 1.0f - cosf(pos0[i]);
    kf0 += 0.5f * m0f[i] * m0f[i];
    kb0 += 0.5f * m0b[i] * m0b[i];
    ppf += 1.0f - cosf(pf[i]);
    kmf += 0.5f * mf[i] * mf[i];
    ppb += 1.0f - cosf(pb[i]);
    kmb += 0.5f * mb[i] * mb[i];
  }
  auto reduce = [&](float v) -> float {
    red[threadIdx.x] = v; __syncthreads();
    for (int s = 128; s; s >>= 1) {
      if (threadIdx.x < s) red[threadIdx.x] += red[threadIdx.x + s];
      __syncthreads();
    }
    float r = red[0]; __syncthreads(); return r;
  };
  p0 = reduce(p0); kf0 = reduce(kf0); kb0 = reduce(kb0);
  ppf = reduce(ppf); kmf = reduce(kmf); ppb = reduce(ppb); kmb = reduce(kmb);

  if (threadIdx.x == 0) {
    float df  = (p0 + kf0) - (ppf + kmf) + sldf[b];
    float db  = (p0 + kb0) - (ppb + kmb) + sldb[b];
    float apf = expf(fminf(df, 0.0f)); if (df != df) apf = 0.0f;
    float apb = expf(fminf(db, 0.0f)); if (db != db) apb = 0.0f;
    float fm  = (u_dir[b] > 0.5f) ? 1.0f : 0.0f;
    float ap  = fm * apf + (1.0f - fm) * apb;
    if (ap != ap) ap = 0.0f;
    float am  = (ap > u_accept[b]) ? 1.0f : 0.0f;
    sh[0] = fm; sh[1] = am;
    out[2 * (size_t)Bn * D + b] = ap;
  }
  __syncthreads();
  const float fm = sh[0], am = sh[1];
  float* pp = out;
  float* mp = out + (size_t)Bn * D;
  float* po = out + 2 * (size_t)Bn * D + Bn;
  for (int d = threadIdx.x; d < D; d += 256) {
    size_t i = (size_t)b * D + d;
    float ppv = fm * pf[i] + (1.0f - fm) * pb[i];
    float mpv = fm * mf[i] + (1.0f - fm) * mb[i];
    pp[i] = ppv;
    mp[i] = mpv;
    po[i] = am * ppv + (1.0f - am) * pos0[i];
  }
}

// ===========================================================================
// Host orchestration
// ===========================================================================
extern "C" void kernel_launch(void* const* d_in, const int* in_sizes, int n_in,
                              void* d_out, int out_size, void* d_ws, size_t ws_size,
                              hipStream_t stream)
{
  (void)in_sizes; (void)n_in; (void)out_size; (void)ws_size;
  constexpr int B = 64, D = 2048, H = 4096, NS = 25;
  constexpr int POS = 0, MOM = 1;
  constexpr int KSPLIT = 8;

  // allow up to 64KB dynamic LDS for the GEMM kernel (idempotent)
  (void)hipFuncSetAttribute((const void*)gemm_bf16_kernel,
                            hipFuncAttributeMaxDynamicSharedMemorySize, 64 * 1024);

  const float* position   = (const float*)d_in[0];
  const float* momentum_f = (const float*)d_in[1];
  const float* momentum_b = (const float*)d_in[2];
  const float* u_dir      = (const float*)d_in[3];
  const float* u_accept   = (const float*)d_in[4];
  const float* eps        = (const float*)d_in[5];
  const float* masks      = (const float*)d_in[6];
  const float* ts         = (const float*)d_in[7];
  const float* W1  = (const float*)d_in[8];
  const float* W2  = (const float*)d_in[9];
  const float* Wt  = (const float*)d_in[10];
  const float* b1  = (const float*)d_in[11];
  const float* Wh  = (const float*)d_in[12];
  const float* bh  = (const float*)d_in[13];
  const float* Ws  = (const float*)d_in[14];
  const float* bs  = (const float*)d_in[15];
  const float* cs  = (const float*)d_in[16];
  const float* Wtr = (const float*)d_in[17];
  const float* btr = (const float*)d_in[18];
  const float* Wq  = (const float*)d_in[19];
  const float* bq  = (const float*)d_in[20];
  const float* cq  = (const float*)d_in[21];

  // ---- workspace carve-up (~250 MB) ----
  char* wsp = (char*)d_ws;
  auto take = [&](size_t bytes) -> char* {
    char* p = wsp; wsp += (bytes + 255) & ~(size_t)255; return p;
  };
  unsigned short* W1p  = (unsigned short*)take((size_t)2 * D * H * 2);
  unsigned short* W2p  = (unsigned short*)take((size_t)2 * D * H * 2);
  unsigned short* Whp  = (unsigned short*)take((size_t)2 * H * H * 2);
  unsigned short* Wsp  = (unsigned short*)take((size_t)2 * H * D * 2);
  unsigned short* Wtrp = (unsigned short*)take((size_t)2 * H * D * 2);
  unsigned short* Wqp  = (unsigned short*)take((size_t)2 * H * D * 2);
  unsigned short* Ap1  = (unsigned short*)take((size_t)B * H * 2);
  unsigned short* Ap2  = (unsigned short*)take((size_t)B * H * 2);
  float* partbuf = (float*)take((size_t)2 * KSPLIT * B * H * 4);   // 16 slots
  float* h0    = (float*)take((size_t)B * H * 4);
  float* hbuf  = (float*)take((size_t)B * H * 4);
  float* yS    = (float*)take((size_t)B * D * 4);
  float* yT    = (float*)take((size_t)B * D * 4);
  float* yQ    = (float*)take((size_t)B * D * 4);
  float* bbuf  = (float*)take((size_t)B * D * 4);
  float* biasM = (float*)take((size_t)H * 4);
  float* biasP = (float*)take((size_t)H * 4);
  float* pos_f  = (float*)take((size_t)B * D * 4);
  float* pos_b  = (float*)take((size_t)B * D * 4);
  float* mom_fs = (float*)take((size_t)B * D * 4);
  float* mom_bs = (float*)take((size_t)B * D * 4);
  float* sldv   = (float*)take((size_t)2 * B * 4);
  float* sldf = sldv;
  float* sldb = sldv + B;

  // ---- one-time (per launch) weight pack: f32 -> bf16 fragment layout ----
  auto packW = [&](const float* src, unsigned short* dst, int K, int N) {
    size_t thr = (size_t)(N / 16) * (K / 32) * 32;
    pack_w_kernel<<<(unsigned)((thr + 255) / 256), 256, 0, stream>>>(src, dst, K, N);
  };
  for (int n = 0; n < 2; ++n) {
    packW(W1  + (size_t)n * D * H, W1p  + (size_t)n * D * H, D, H);
    packW(W2  + (size_t)n * D * H, W2p  + (size_t)n * D * H, D, H);
    packW(Wh  + (size_t)n * H * H, Whp  + (size_t)n * H * H, H, H);
    packW(Ws  + (size_t)n * H * D, Wsp  + (size_t)n * H * D, H, D);
    packW(Wtr + (size_t)n * H * D, Wtrp + (size_t)n * H * D, H, D);
    packW(Wq  + (size_t)n * H * D, Wqp  + (size_t)n * H * D, H, D);
  }

  // ---- state init ----
  hipMemcpyAsync(pos_f,  position,   (size_t)B * D * 4, hipMemcpyDeviceToDevice, stream);
  hipMemcpyAsync(pos_b,  position,   (size_t)B * D * 4, hipMemcpyDeviceToDevice, stream);
  hipMemcpyAsync(mom_fs, momentum_f, (size_t)B * D * 4, hipMemcpyDeviceToDevice, stream);
  hipMemcpyAsync(mom_bs, momentum_b, (size_t)B * D * 4, hipMemcpyDeviceToDevice, stream);
  hipMemsetAsync(sldv, 0, (size_t)2 * B * 4, stream);

  auto packA = [&](const float* src, unsigned short* dst, int K) {
    int thr = 4 * (K >> 5) * 32;
    pack_a_kernel<<<(thr + 255) / 256, 256, 0, stream>>>(src, dst, K);
  };

  // partial-sum GEMM launch: grid (N/128, KSPLIT), dyn-LDS = A chunk bytes
  auto gemm = [&](const unsigned short* a, const unsigned short* w,
                  int slotOff, int N_, int K_) {
    int KT  = K_ >> 5;
    int KTB = KT / KSPLIT;
    size_t smem = (size_t)4 * KTB * 1024;
    dim3 grid(N_ / 128, KSPLIT);
    gemm_bf16_kernel<<<grid, 256, smem, stream>>>(a, w, partbuf, slotOff, N_, KT, KTB);
  };

  // ---- 3-layer net given packed inputs in Ap1/Ap2 (K=D) ----
  auto run_layers = [&](int nn, const float* bias1) {
    // layer 1: h0 = relu(A@W1 + G@W2 + bias1)   (two independent partial GEMMs)
    gemm(Ap1, W1p + (size_t)nn * D * H, 0,      H, D);
    gemm(Ap2, W2p + (size_t)nn * D * H, KSPLIT, H, D);
    epilogue_kernel<true, true><<<(B * H) / 256, 256, 0, stream>>>(
        partbuf, 2 * KSPLIT, H, bias1, h0, B * H);
    packA(h0, Ap1, H);
    // layer 2: h = relu(h0@Wh + bh)
    gemm(Ap1, Whp + (size_t)nn * H * H, 0, H, H);
    epilogue_kernel<true, true><<<(B * H) / 256, 256, 0, stream>>>(
        partbuf, KSPLIT, H, bh + (size_t)nn * H, hbuf, B * H);
    packA(hbuf, Ap1, H);
    // layer 3 heads (raw y; bias/tanh/exp fused into update kernels)
    gemm(Ap1, Wsp + (size_t)nn * H * D, 0, D, H);
    epilogue_kernel<false, false><<<(B * D) / 256, 256, 0, stream>>>(
        partbuf, KSPLIT, D, nullptr, yS, B * D);
    gemm(Ap1, Wtrp + (size_t)nn * H * D, 0, D, H);
    epilogue_kernel<false, false><<<(B * D) / 256, 256, 0, stream>>>(
        partbuf, KSPLIT, D, nullptr, yT, B * D);
    gemm(Ap1, Wqp + (size_t)nn * H * D, 0, D, H);
    epilogue_kernel<false, false><<<(B * D) / 256, 256, 0, stream>>>(
        partbuf, KSPLIT, D, nullptr, yQ, B * D);
  };

  auto mom_step = [&](float* pos, float* mom, float* sld, int backward) {
    grad_kernel<<<(B * D) / 256, 256, 0, stream>>>(pos, bbuf, B * D);   // g = sin(pos)
    packA(pos,  Ap1, D);
    packA(bbuf, Ap2, D);
    run_layers(MOM, biasM);
    mom_update_kernel<<<B, 256, 0, stream>>>(
        mom, bbuf, yS, yT, yQ,
        bs + (size_t)MOM * D, cs + (size_t)MOM * D, btr + (size_t)MOM * D,
        bq + (size_t)MOM * D, cq + (size_t)MOM * D,
        eps, sld, D, backward);
  };

  auto pos_step = [&](float* pos, float* mom, float* sld, const float* mask,
                      int swap, int backward) {
    maskmul_kernel<<<(B * D) / 256, 256, 0, stream>>>(pos, mask, bbuf, D, B * D, swap);
    packA(mom,  Ap1, D);   // a = mom
    packA(bbuf, Ap2, D);   // b = m * pos
    run_layers(POS, biasP);
    pos_update_kernel<<<B, 256, 0, stream>>>(
        pos, mom, mask, yS, yT, yQ,
        bs + (size_t)POS * D, cs + (size_t)POS * D, btr + (size_t)POS * D,
        bq + (size_t)POS * D, cq + (size_t)POS * D,
        eps, sld, D, swap, backward);
  };

  // ---- forward chain ----
  for (int i = 0; i < NS; ++i) {
    const float* mask = masks + (size_t)i * D;
    bias1_kernel<<<H / 256, 256, 0, stream>>>(b1 + (size_t)MOM * H, Wt + (size_t)MOM * 2 * H, ts, i, biasM, H);
    bias1_kernel<<<H / 256, 256, 0, stream>>>(b1 + (size_t)POS * H, Wt + (size_t)POS * 2 * H, ts, i, biasP, H);
    mom_step(pos_f, mom_fs, sldf, 0);
    pos_step(pos_f, mom_fs, sldf, mask, 0, 0);   // (m, 1-m)
    pos_step(pos_f, mom_fs, sldf, mask, 1, 0);   // (1-m, m)
    mom_step(pos_f, mom_fs, sldf, 0);
  }

  // ---- backward chain (reversed masks/ts) ----
  for (int i = 0; i < NS; ++i) {
    int si = NS - 1 - i;
    const float* mask = masks + (size_t)si * D;
    bias1_kernel<<<H / 256, 256, 0, stream>>>(b1 + (size_t)MOM * H, Wt + (size_t)MOM * 2 * H, ts, si, biasM, H);
    bias1_kernel<<<H / 256, 256, 0, stream>>>(b1 + (size_t)POS * H, Wt + (size_t)POS * 2 * H, ts, si, biasP, H);
    mom_step(pos_b, mom_bs, sldb, 1);
    pos_step(pos_b, mom_bs, sldb, mask, 1, 1);   // (1-m, m)
    pos_step(pos_b, mom_bs, sldb, mask, 0, 1);   // (m, 1-m)
    mom_step(pos_b, mom_bs, sldb, 1);
  }

  // ---- accept / combine ----
  finalize_kernel<<<B, 256, 0, stream>>>(
      position, momentum_f, momentum_b,
      pos_f, mom_fs, pos_b, mom_bs,
      sldf, sldb, u_dir, u_accept, (float*)d_out, B, D);
}
